// BitVocoder_68118181314700
// MI455X (gfx1250) — matrix operand order repaired and checked by reading, not compiled, not run on hardware
//
#include <hip/hip_runtime.h>
#include <math.h>

typedef __bf16 bf16x;
typedef __attribute__((ext_vector_type(16))) __bf16 v16bf;
typedef __attribute__((ext_vector_type(8)))  __bf16 v8bf;
typedef __attribute__((ext_vector_type(8)))  float   v8f;

// ------------------------------------------------------------------
// absmean scale for BitNet quantization (deterministic single-block)
static __global__ void k_absmean(const float* __restrict__ w, int n,
                                 float* __restrict__ scale) {
  __shared__ float sm[1024];
  float s = 0.f;
  for (int i = threadIdx.x; i < n; i += 1024) s += fabsf(w[i]);
  sm[threadIdx.x] = s;
  __syncthreads();
  for (int off = 512; off > 0; off >>= 1) {
    if ((int)threadIdx.x < off) sm[threadIdx.x] += sm[threadIdx.x + off];
    __syncthreads();
  }
  if (threadIdx.x == 0) *scale = sm[0] / (float)n + 1e-5f;
}

// quantize (optional) + repack (O,I,K) -> bf16 [Op, k*I + i], rows >= O zeroed.
// Output rows padded to Op (multiple of 64) so the GEMM needs no edge masking.
static __global__ void k_quant_repack(const float* __restrict__ w,
                                      bf16x* __restrict__ out,
                                      const float* __restrict__ scale,
                                      int O, int Op, int I, int K, int do_quant) {
  long idx = (long)blockIdx.x * blockDim.x + threadIdx.x;
  long total = (long)Op * I * K;
  if (idx >= total) return;
  // output layout: idx = (o*K + k)*I + i
  int i = (int)(idx % I);
  long r = idx / I;
  int k = (int)(r % K);
  int o = (int)(r / K);
  float v = 0.f;
  if (o < O) {
    v = w[((size_t)o * I + i) * K + k];
    if (do_quant) {
      float s = *scale;
      v = fminf(fmaxf(rintf(v / s), -1.f), 1.f) * s;
    }
  }
  out[idx] = (bf16x)v;
}

// im2col -> bf16 A matrix [M, taps*Cin], cols ordered k*Cin + c ('same' pad)
// bct=1: src is (B, Cin, T) (the network input); bct=0: src is (B, T, Cin)
static __global__ void k_im2col(const float* __restrict__ src,
                                bf16x* __restrict__ dst,
                                int Bn, int T, int Cin, int taps, int dil, int bct) {
  long idx = (long)blockIdx.x * blockDim.x + threadIdx.x;
  long total = (long)Bn * T * taps * Cin;
  if (idx >= total) return;
  int c = (int)(idx % Cin);
  long r = idx / Cin;
  int k = (int)(r % taps);
  long row = r / taps;
  int t = (int)(row % T);
  int b = (int)(row / T);
  int tt = t + dil * (k - taps / 2);
  float v = 0.f;
  if (tt >= 0 && tt < T)
    v = bct ? src[((size_t)b * Cin + c) * T + tt]
            : src[((size_t)b * T + tt) * Cin + c];
  dst[idx] = (bf16x)v;
}

// ------------------------------------------------------------------
template <int K>
__device__ __forceinline__ v16bf load_afrag(const bf16x* __restrict__ Ar,
                                            int k0, int g) {
  // A fragment: K chunks [k0+16g, +8) and [k0+16g+8, +8) (contiguous, matches B)
  v8bf a0 = *(const v8bf*)(Ar + k0 + 16 * g);
  v8bf a1 = *(const v8bf*)(Ar + k0 + 16 * g + 8);
  return __builtin_shufflevector(a0, a1, 0, 1, 2, 3, 4, 5, 6, 7,
                                 8, 9, 10, 11, 12, 13, 14, 15);
}

// WMMA GEMM: C[M,N] = A[M,K]bf16 @ W[Np,K]^T bf16 + bias[N]  (f32 accum)
// K is a compile-time constant -> all tile strides fold into load immediates,
// eliminating per-iteration 64-bit address VALU and its WMMA hazards.
// W row-padded to Np (multiple of 64, padded rows zero): no edge masking.
// Each wave owns a 32x64 output tile: two M-tiles share the four B fragments,
// so each 32-K chunk is 12 loads for 8 WMMAs (vs 10 for 4 with one M-tile).
// Fragments are software-pipelined (next chunk's loads issued before the
// current 8 WMMAs) so the VMEM pipe overlaps the matrix pipe.
// 8 waves per 256-thread block; grid.y covers M/32 row bands.
// NOTE: all K here are multiples of 64; the final-iteration lookahead reads
// <=64B past the row end, which stays inside the (tail-padded) workspace.
template <int K>
static __global__ void k_gemm(const bf16x* __restrict__ A,
                              const bf16x* __restrict__ W,
                              float* __restrict__ C,
                              const float* __restrict__ bias,
                              int N, int ldc, int ngrp) {
  const int lane = threadIdx.x & 31;
  const int wave = threadIdx.x >> 5;
  const int mt = blockIdx.y;                 // 32-row band
  const int grp = blockIdx.x * 8 + wave;     // group of 4 N-tiles
  if (grp >= ngrp) return;                   // wave-uniform exit
  const int nt0 = grp * 4;
  const int g = lane >> 4;
  const int l16 = lane & 15;
  const bf16x* Ar = A + (size_t)(mt * 32 + l16) * K;  // M-tile0 row
  const bf16x* Wr = W + (size_t)(nt0 * 16 + l16) * K;
  constexpr size_t WS = (size_t)16 * K;      // stride between N-tiles
  constexpr size_t AS = (size_t)16 * K;      // stride to second M-tile

  // fold bias into accumulator init (per-lane output column is fixed)
  const int cn0 = nt0 * 16 + l16;
  float b0 = (cn0      < N) ? bias[cn0]      : 0.f;
  float b1 = (cn0 + 16 < N) ? bias[cn0 + 16] : 0.f;
  float b2 = (cn0 + 32 < N) ? bias[cn0 + 32] : 0.f;
  float b3 = (cn0 + 48 < N) ? bias[cn0 + 48] : 0.f;
  v8f c00 = {b0, b0, b0, b0, b0, b0, b0, b0};
  v8f c01 = {b1, b1, b1, b1, b1, b1, b1, b1};
  v8f c02 = {b2, b2, b2, b2, b2, b2, b2, b2};
  v8f c03 = {b3, b3, b3, b3, b3, b3, b3, b3};
  v8f c10 = c00, c11 = c01, c12 = c02, c13 = c03;

  // ---- software pipeline: prologue loads for k0 = 0
  v16bf a0C = load_afrag<K>(Ar, 0, g);
  v16bf a1C = load_afrag<K>(Ar + AS, 0, g);
  v16bf b0C = *(const v16bf*)(Wr + 16 * g);
  v16bf b1C = *(const v16bf*)(Wr + WS + 16 * g);
  v16bf b2C = *(const v16bf*)(Wr + 2 * WS + 16 * g);
  v16bf b3C = *(const v16bf*)(Wr + 3 * WS + 16 * g);

#pragma unroll 2
  for (int k0 = 0; k0 < K; k0 += 32) {
    const int kn = k0 + 32;                  // lookahead chunk
    v16bf a0N = load_afrag<K>(Ar, kn, g);
    v16bf a1N = load_afrag<K>(Ar + AS, kn, g);
    v16bf b0N = *(const v16bf*)(Wr + kn + 16 * g);
    v16bf b1N = *(const v16bf*)(Wr + WS + kn + 16 * g);
    v16bf b2N = *(const v16bf*)(Wr + 2 * WS + kn + 16 * g);
    v16bf b3N = *(const v16bf*)(Wr + 3 * WS + kn + 16 * g);
    __builtin_prefetch((const void*)(Ar + k0 + 512), 0, 3);
    c00 = __builtin_amdgcn_wmma_f32_16x16x32_bf16(false, a0C, false, b0C,
                                                  (short)0, c00, false, false);
    c10 = __builtin_amdgcn_wmma_f32_16x16x32_bf16(false, a1C, false, b0C,
                                                  (short)0, c10, false, false);
    c01 = __builtin_amdgcn_wmma_f32_16x16x32_bf16(false, a0C, false, b1C,
                                                  (short)0, c01, false, false);
    c11 = __builtin_amdgcn_wmma_f32_16x16x32_bf16(false, a1C, false, b1C,
                                                  (short)0, c11, false, false);
    c02 = __builtin_amdgcn_wmma_f32_16x16x32_bf16(false, a0C, false, b2C,
                                                  (short)0, c02, false, false);
    c12 = __builtin_amdgcn_wmma_f32_16x16x32_bf16(false, a1C, false, b2C,
                                                  (short)0, c12, false, false);
    c03 = __builtin_amdgcn_wmma_f32_16x16x32_bf16(false, a0C, false, b3C,
                                                  (short)0, c03, false, false);
    c13 = __builtin_amdgcn_wmma_f32_16x16x32_bf16(false, a1C, false, b3C,
                                                  (short)0, c13, false, false);
    a0C = a0N; a1C = a1N; b0C = b0N; b1C = b1N; b2C = b2N; b3C = b3N;
  }

  // C/D layout: VGPR r -> row base + r + 8*(lane/16), col = tile*16 + lane%16
  const int rb = mt * 32 + 8 * g;
#pragma unroll
  for (int r = 0; r < 8; ++r) {
    float* Crow0 = C + (size_t)(rb + r) * ldc;        // M-tile0
    float* Crow1 = C + (size_t)(rb + 16 + r) * ldc;   // M-tile1
    if (cn0      < N) { Crow0[cn0]      = c00[r]; Crow1[cn0]      = c10[r]; }
    if (cn0 + 16 < N) { Crow0[cn0 + 16] = c01[r]; Crow1[cn0 + 16] = c11[r]; }
    if (cn0 + 32 < N) { Crow0[cn0 + 32] = c02[r]; Crow1[cn0 + 32] = c12[r]; }
    if (cn0 + 48 < N) { Crow0[cn0 + 48] = c03[r]; Crow1[cn0 + 48] = c13[r]; }
  }
}

// ------------------------------------------------------------------
// SnakeBeta: x + 1/(b+eps) * sin(a*x)^2, channel = idx % Cmod
static __global__ void k_snake(float* __restrict__ x, const float* __restrict__ a,
                               const float* __restrict__ b, int Cmod, long n) {
  long idx = (long)blockIdx.x * blockDim.x + threadIdx.x;
  if (idx >= n) return;
  int c = (int)(idx % Cmod);
  float xv = x[idx];
  float s = sinf(a[c] * xv);
  x[idx] = xv + (1.f / (b[c] + 1e-9f)) * s * s;
}

// RMSNorm in place, one block (256 thr) per row of length C
static __global__ void k_rmsnorm(float* __restrict__ x, const float* __restrict__ w,
                                 int C) {
  __shared__ float sm[256];
  __shared__ float rinv;
  float* p = x + (size_t)blockIdx.x * C;
  float s = 0.f;
  for (int c = threadIdx.x; c < C; c += 256) { float v = p[c]; s += v * v; }
  sm[threadIdx.x] = s;
  __syncthreads();
  for (int off = 128; off > 0; off >>= 1) {
    if ((int)threadIdx.x < off) sm[threadIdx.x] += sm[threadIdx.x + off];
    __syncthreads();
  }
  if (threadIdx.x == 0) rinv = rsqrtf(sm[0] / (float)C + 1e-6f);
  __syncthreads();
  for (int c = threadIdx.x; c < C; c += 256) p[c] = p[c] * rinv * w[c];
}

// depthwise 7-tap dilated conv over time, (B,T,C) layout. w: [C,7]
static __global__ void k_dw7(const float* __restrict__ h, const float* __restrict__ w,
                             const float* __restrict__ bias, float* __restrict__ y,
                             int Bn, int T, int C, int dil) {
  long idx = (long)blockIdx.x * blockDim.x + threadIdx.x;
  long n = (long)Bn * T * C;
  if (idx >= n) return;
  int c = (int)(idx % C);
  long row = idx / C;
  int t = (int)(row % T);
  int b = (int)(row / T);
  float acc = bias[c];
#pragma unroll
  for (int k = 0; k < 7; ++k) {
    int tt = t + dil * (k - 3);
    if (tt >= 0 && tt < T) acc += w[c * 7 + k] * h[((size_t)b * T + tt) * C + c];
  }
  y[idx] = acc;
}

static __global__ void k_axpy_gamma(float* __restrict__ h, const float* __restrict__ y,
                                    const float* __restrict__ gamma, int C, long n) {
  long idx = (long)blockIdx.x * blockDim.x + threadIdx.x;
  if (idx >= n) return;
  h[idx] += gamma[idx % C] * y[idx];
}

static __global__ void k_add(float* __restrict__ h, const float* __restrict__ s, long n) {
  long idx = (long)blockIdx.x * blockDim.x + threadIdx.x;
  if (idx >= n) return;
  h[idx] += s[idx];
}

static __global__ void k_expclip(const float* __restrict__ x, float* __restrict__ out,
                                 long n) {
  long idx = (long)blockIdx.x * blockDim.x + threadIdx.x;
  if (idx >= n) return;
  out[idx] = expf(fminf(fmaxf(x[idx], -10.f), 10.f));
}

static __global__ void k_softmax4(const float* __restrict__ w, float* __restrict__ out) {
  if (threadIdx.x == 0 && blockIdx.x == 0) {
    float m = fmaxf(fmaxf(w[0], w[1]), fmaxf(w[2], w[3]));
    float e0 = expf(w[0] - m), e1 = expf(w[1] - m), e2 = expf(w[2] - m),
          e3 = expf(w[3] - m);
    float s = e0 + e1 + e2 + e3;
    out[0] = e0 / s; out[1] = e1 / s; out[2] = e2 / s; out[3] = e3 / s;
  }
}

// head-mix (softmax over 4 heads) + tanh*0.95pi*cumsum_w -> (M, OUTC)
static __global__ void k_headmix(const float* __restrict__ f, const float* __restrict__ sm,
                                 const float* __restrict__ cw, float* __restrict__ out,
                                 long M, int OUTC) {
  long idx = (long)blockIdx.x * blockDim.x + threadIdx.x;
  long n = M * OUTC;
  if (idx >= n) return;
  int o = (int)(idx % OUTC);
  long row = idx / OUTC;
  const float* fr = f + row * (size_t)(4 * OUTC);
  float acc = fr[o] * sm[0] + fr[OUTC + o] * sm[1] + fr[2 * OUTC + o] * sm[2] +
              fr[3 * OUTC + o] * sm[3];
  out[idx] = tanhf(acc) * (3.14159265358979323846f * 0.95f) * cw[o];
}

// cumulative sum over t per (b, freq) + phase_bias
static __global__ void k_cumsum(const float* __restrict__ f, const float* __restrict__ pb,
                                float* __restrict__ out, int Bn, int T, int OUTC) {
  long tid = (long)blockIdx.x * blockDim.x + threadIdx.x;
  long n = (long)Bn * OUTC;
  if (tid >= n) return;
  int o = (int)(tid % OUTC);
  int b = (int)(tid / OUTC);
  float acc = 0.f;
  for (int t = 0; t < T; ++t) {
    size_t i = ((size_t)b * T + t) * OUTC + o;
    acc += f[i];
    out[i] = acc + pb[o];
  }
}

// 5-tap temporal smoothing (depthwise over time, shared kernel)
static __global__ void k_smooth5(const float* __restrict__ x, const float* __restrict__ sm,
                                 float* __restrict__ out, int Bn, int T, int OUTC) {
  long idx = (long)blockIdx.x * blockDim.x + threadIdx.x;
  long n = (long)Bn * T * OUTC;
  if (idx >= n) return;
  int o = (int)(idx % OUTC);
  long row = idx / OUTC;
  int t = (int)(row % T);
  int b = (int)(row / T);
  float acc = 0.f;
#pragma unroll
  for (int j = 0; j < 5; ++j) {
    int tt = t + j - 2;
    if (tt >= 0 && tt < T) acc += sm[j] * x[((size_t)b * T + tt) * OUTC + o];
  }
  out[idx] = acc;
}

static __global__ void k_wrap(const float* __restrict__ x, float* __restrict__ out,
                              long n) {
  long idx = (long)blockIdx.x * blockDim.x + threadIdx.x;
  if (idx >= n) return;
  float v = x[idx];
  out[idx] = atan2f(sinf(v), cosf(v));
}

// ------------------------------------------------------------------
extern "C" void kernel_launch(void* const* d_in, const int* in_sizes, int n_in,
                              void* d_out, int out_size, void* d_ws, size_t ws_size,
                              hipStream_t stream) {
  (void)in_sizes; (void)n_in; (void)out_size; (void)ws_size;
  constexpr int Bn = 4, T = 3000, INC = 256, Cd = 512, EXPD = 2048, OUTC = 513;
  constexpr int M = Bn * T;               // 12000 rows (multiple of 32)
  constexpr int DILS[6] = {1, 2, 4, 8, 16, 32};
  constexpr int RDILS[3] = {1, 3, 5};

  auto in = [&](int i) { return (const float*)d_in[i]; };
  const float* x      = in(0);
  const float* skip_w = in(1);  const float* skip_b = in(2);
  const float* in_w   = in(3);  const float* in_b   = in(4);
  const float* in_a   = in(5);  const float* in_bt  = in(6);
  // blocks: 7 + blk*10 : dw_w,dw_b,norm_w,pw1_w,pw1_b,a1,b1,pw2_w,pw2_b,gamma
  // res:    67 + (stack*3+layer)*6 : c1_w,c1_b,a,b,c2_w,c2_b
  const float* final_norm_w = in(103);
  const float* m_bl_w = in(104); const float* m_bl_b = in(105);
  const float* m_a = in(106);    const float* m_b = in(107);
  const float* m_rms_w = in(108);
  const float* m_lin_w = in(109); const float* m_lin_b = in(110);
  const float* p_rms_w = in(111);
  const float* p_lin_w = in(112); const float* p_lin_b = in(113);
  const float* p_cw = in(114);    const float* p_bias = in(115);
  const float* p_head_w = in(116); const float* p_smooth = in(117);

  char* ws = (char*)d_ws;
  size_t off = 0;
  auto alloc = [&](size_t bytes) {
    size_t o = off; off = (off + bytes + 255) & ~(size_t)255; return o;
  };
  float* h    = (float*)(ws + alloc((size_t)M * Cd * 4));
  float* skp  = (float*)(ws + alloc((size_t)M * OUTC * 4)); // skip, later phase buf
  float* tmp  = (float*)(ws + alloc((size_t)M * 2052 * 4));
  bf16x* Abuf = (bf16x*)(ws + alloc((size_t)M * 2048 * 2));
  bf16x* Wbuf = (bf16x*)(ws + alloc((size_t)4u << 20));     // 2112*512*2 max fits
  float* y32  = (float*)(ws + alloc((size_t)M * Cd * 4));
  float* scal = (float*)(ws + alloc(256));
  float* smx  = scal + 8;
  (void)alloc(4096);  // tail pad: GEMM lookahead may read <=64B past last row

  auto eg = [&](long n) { return dim3((unsigned)((n + 255) / 256)); };
  auto padN = [](int n) { return (n + 63) & ~63; };
  auto quantrepack = [&](const float* w, int O, int I, int K) {
    k_absmean<<<1, 1024, 0, stream>>>(w, O * I * K, scal);
    int Op = padN(O);
    k_quant_repack<<<eg((long)Op * I * K), 256, 0, stream>>>(w, Wbuf, scal,
                                                             O, Op, I, K, 1);
  };
  auto plainw = [&](const float* w, int O, int I) {
    int Op = padN(O);
    k_quant_repack<<<eg((long)Op * I), 256, 0, stream>>>(w, Wbuf, scal,
                                                         O, Op, I, 1, 0);
  };
  auto im2col = [&](const float* src, int Cin, int taps, int dil, int bct) {
    k_im2col<<<eg((long)M * taps * Cin), 256, 0, stream>>>(src, Abuf, Bn, T, Cin,
                                                           taps, dil, bct);
  };
  auto gemm = [&](float* C, const float* bias, int N, int K, int ldc) {
    int ngrp = padN(N) / 64;                 // 4 N-tiles (64 cols) per wave
    dim3 grid((unsigned)((ngrp + 7) / 8), M / 32);   // 32-row bands
    switch (K) {                             // compile-time K instantiations
      case 256:
        k_gemm<256><<<grid, 256, 0, stream>>>(Abuf, Wbuf, C, bias, N, ldc, ngrp);
        break;
      case 512:
        k_gemm<512><<<grid, 256, 0, stream>>>(Abuf, Wbuf, C, bias, N, ldc, ngrp);
        break;
      case 1536:
        k_gemm<1536><<<grid, 256, 0, stream>>>(Abuf, Wbuf, C, bias, N, ldc, ngrp);
        break;
      case 1792:
        k_gemm<1792><<<grid, 256, 0, stream>>>(Abuf, Wbuf, C, bias, N, ldc, ngrp);
        break;
      case 2048:
        k_gemm<2048><<<grid, 256, 0, stream>>>(Abuf, Wbuf, C, bias, N, ldc, ngrp);
        break;
      default: break;                        // unreachable for this network
    }
  };

  // ---- skip = bitq(skip_w) * x  (1x1 conv, K=256)
  quantrepack(skip_w, Cd, INC, 1);
  im2col(x, INC, 1, 1, /*bct=*/1);
  gemm(skp, skip_b, Cd, INC, Cd);

  // ---- h = snake(conv7(x)) (256->512, K=1792)
  quantrepack(in_w, Cd, INC, 7);
  im2col(x, INC, 7, 1, /*bct=*/1);
  gemm(h, in_b, Cd, 7 * INC, Cd);
  k_snake<<<eg((long)M * Cd), 256, 0, stream>>>(h, in_a, in_bt, Cd, (long)M * Cd);

  // ---- 6 ConvNeXt blocks
  for (int blk = 0; blk < 6; ++blk) {
    int base = 7 + blk * 10;
    const float* dw_w = in(base + 0); const float* dw_b = in(base + 1);
    const float* nw   = in(base + 2);
    const float* pw1w = in(base + 3); const float* pw1b = in(base + 4);
    const float* a1   = in(base + 5); const float* b1   = in(base + 6);
    const float* pw2w = in(base + 7); const float* pw2b = in(base + 8);
    const float* gam  = in(base + 9);
    int d = DILS[blk];
    k_dw7<<<eg((long)M * Cd), 256, 0, stream>>>(h, dw_w, dw_b, y32, Bn, T, Cd, d);
    k_rmsnorm<<<M, 256, 0, stream>>>(y32, nw, Cd);
    quantrepack(pw1w, EXPD, Cd, 1);
    im2col(y32, Cd, 1, 1, 0);
    gemm(tmp, pw1b, EXPD, Cd, EXPD);
    k_snake<<<eg((long)M * EXPD), 256, 0, stream>>>(tmp, a1, b1, EXPD, (long)M * EXPD);
    quantrepack(pw2w, Cd, EXPD, 1);
    im2col(tmp, EXPD, 1, 1, 0);
    gemm(y32, pw2b, Cd, EXPD, Cd);
    k_axpy_gamma<<<eg((long)M * Cd), 256, 0, stream>>>(h, y32, gam, Cd, (long)M * Cd);
  }

  // ---- h += skip
  k_add<<<eg((long)M * Cd), 256, 0, stream>>>(h, skp, (long)M * Cd);

  // ---- residual conv stacks (2 x 3 layers, dil 1/3/5), K=1536 GEMMs
  for (int st = 0; st < 2; ++st) {
    for (int ly = 0; ly < 3; ++ly) {
      int base = 67 + (st * 3 + ly) * 6;
      const float* c1w = in(base + 0); const float* c1b = in(base + 1);
      const float* aa  = in(base + 2); const float* bb  = in(base + 3);
      const float* c2w = in(base + 4); const float* c2b = in(base + 5);
      int d = RDILS[ly];
      quantrepack(c1w, Cd, Cd, 3);
      im2col(h, Cd, 3, d, 0);
      gemm(y32, c1b, Cd, 3 * Cd, Cd);
      k_snake<<<eg((long)M * Cd), 256, 0, stream>>>(y32, aa, bb, Cd, (long)M * Cd);
      quantrepack(c2w, Cd, Cd, 3);
      im2col(y32, Cd, 3, 1, 0);
      gemm(tmp, c2b, Cd, 3 * Cd, Cd);
      k_add<<<eg((long)M * Cd), 256, 0, stream>>>(h, tmp, (long)M * Cd);
    }
  }

  // ---- final RMSNorm
  k_rmsnorm<<<M, 256, 0, stream>>>(h, final_norm_w, Cd);

  float* out_mag = (float*)d_out;
  float* out_ph  = out_mag + (size_t)M * OUTC;

  // ---- magnitude head
  quantrepack(m_bl_w, Cd, Cd, 1);
  im2col(h, Cd, 1, 1, 0);
  gemm(y32, m_bl_b, Cd, Cd, Cd);
  k_snake<<<eg((long)M * Cd), 256, 0, stream>>>(y32, m_a, m_b, Cd, (long)M * Cd);
  k_rmsnorm<<<M, 256, 0, stream>>>(y32, m_rms_w, Cd);
  plainw(m_lin_w, OUTC, Cd);
  im2col(y32, Cd, 1, 1, 0);
  gemm(tmp, m_lin_b, OUTC, Cd, OUTC);
  k_expclip<<<eg((long)M * OUTC), 256, 0, stream>>>(tmp, out_mag, (long)M * OUTC);

  // ---- phase head
  k_rmsnorm<<<M, 256, 0, stream>>>(h, p_rms_w, Cd);     // h no longer needed
  plainw(p_lin_w, 4 * OUTC, Cd);
  im2col(h, Cd, 1, 1, 0);
  gemm(tmp, p_lin_b, 4 * OUTC, Cd, 4 * OUTC);
  k_softmax4<<<1, 32, 0, stream>>>(p_head_w, smx);
  k_headmix<<<eg((long)M * OUTC), 256, 0, stream>>>(tmp, smx, p_cw, skp, (long)M, OUTC);
  k_cumsum<<<eg((long)Bn * OUTC), 256, 0, stream>>>(skp, p_bias, tmp, Bn, T, OUTC);
  k_smooth5<<<eg((long)M * OUTC), 256, 0, stream>>>(tmp, p_smooth, skp, Bn, T, OUTC);
  k_wrap<<<eg((long)M * OUTC), 256, 0, stream>>>(skp, out_ph, (long)M * OUTC);
}
